// Decoder_88261577932983
// MI455X (gfx1250) — compile-verified, matrix operand order checked by reading, not run on hardware
//
#include <hip/hip_runtime.h>
#include <hip/hip_bf16.h>

// ---------------------------------------------------------------------------
// DA-RNN decoder for MI455X (gfx1250, wave32, WMMA).
//  - bf16 mirrors of the two big streamed arrays (U, input_encoded) halve the
//    per-step HBM traffic (23.3 TB/s roofline -> ~1.45 ms floor for the scan).
//  - All GEMMs on v_wmma_f32_16x16x32_bf16, 16x64 tile per wave (1 A-frag
//    reused across 4 B-frags / 4 accumulators).
//  - Attention kernel streams with global_load_b64 (4 x bf16 per lane) and
//    global_prefetch one row ahead.
// ---------------------------------------------------------------------------

typedef __attribute__((ext_vector_type(16))) __bf16 v16bf;
typedef __attribute__((ext_vector_type(4)))  __bf16 bf16x4;
typedef __attribute__((ext_vector_type(8)))  float  v8f;

#define B_  4096
#define T1_ 127
#define E_  128
#define D_  128

union FragU { v16bf f; float4 q[2]; };

// CDNA5 16-bit A/B fragment layout (ISA 7.12.2): a lane's 16 elements are two
// contiguous 8-element K-runs -> two global_load_b128 per fragment.
__device__ __forceinline__ v16bf load_frag(const __bf16* rowbase, int kblock, int half) {
    FragU u;
    const __bf16* p = rowbase + kblock + half * 8;
    u.q[0] = *(const float4*)(p);        // k = kblock + half*8 .. +7
    u.q[1] = *(const float4*)(p + 16);   // k = kblock + 16 + half*8 .. +7
    return u.f;
}

// Wave computes a 16(M) x 64(N) tile: one A fragment feeds 4 WMMAs per K-step.
// D[m,n] = sum_k A[m,k]*W[n,k] (+ colbias[n]) (+ rowvec[m]*colvec[n]).
__global__ void wmma_gemm4(const __bf16* __restrict__ A, const __bf16* __restrict__ W,
                           const float* __restrict__ colbias,
                           const float* __restrict__ rowvec, const float* __restrict__ colvec,
                           float* __restrict__ Df, __bf16* __restrict__ Dbf,
                           int M, int N, int K)
{
    int wave   = (int)((blockIdx.x * blockDim.x + threadIdx.x) >> 5);
    int lane   = threadIdx.x & 31;
    int tilesN = N >> 6;                        // 64-wide N tiles
    int mt  = wave / tilesN;
    int nt4 = wave % tilesN;
    if (mt * 16 >= M) return;                   // wave-uniform: EXEC all-1s for WMMA
    int half = lane >> 4;
    int sub  = lane & 15;

    const __bf16* arow = A + (size_t)(mt * 16 + sub) * K;
    const __bf16* brow = W + (size_t)(nt4 * 64 + sub) * K;
    const size_t  bstep = (size_t)16 * K;

    v8f acc0 = {}, acc1 = {}, acc2 = {}, acc3 = {};
    for (int kb = 0; kb < K; kb += 32) {
        v16bf af  = load_frag(arow, kb, half);
        v16bf bf0 = load_frag(brow,             kb, half);
        v16bf bf1 = load_frag(brow + bstep,     kb, half);
        v16bf bf2 = load_frag(brow + 2 * bstep, kb, half);
        v16bf bf3 = load_frag(brow + 3 * bstep, kb, half);
        acc0 = __builtin_amdgcn_wmma_f32_16x16x32_bf16(false, af, false, bf0, (short)0, acc0, false, false);
        acc1 = __builtin_amdgcn_wmma_f32_16x16x32_bf16(false, af, false, bf1, (short)0, acc1, false, false);
        acc2 = __builtin_amdgcn_wmma_f32_16x16x32_bf16(false, af, false, bf2, (short)0, acc2, false, false);
        acc3 = __builtin_amdgcn_wmma_f32_16x16x32_bf16(false, af, false, bf3, (short)0, acc3, false, false);
    }

    v8f accs[4] = {acc0, acc1, acc2, acc3};
#pragma unroll
    for (int q = 0; q < 4; ++q) {
        int n = nt4 * 64 + q * 16 + sub;
        float cb = colbias ? colbias[n] : 0.f;
        float cv = (rowvec && colvec) ? colvec[n] : 0.f;
#pragma unroll
        for (int r = 0; r < 8; ++r) {
            int m = mt * 16 + r + half * 8;     // C/D layout: vgpr r, halves -> m
            float v = accs[q][r] + cb;
            if (rowvec && colvec) v += rowvec[m] * cv;
            if (Df)  Df [(size_t)m * N + n] = v;
            if (Dbf) Dbf[(size_t)m * N + n] = (__bf16)v;
        }
    }
}

// Fused attention: scores = v . tanh(proj + U[b,t1,:]) ; softmax over T1 ;
// ctx = sum_t beta_t * enc[b,t,:] ; y_tilde = [ctx, y_t] @ w^T + b.
// One 128-thread block (4 waves) per batch row; 4 bf16 per lane per load.
__global__ void attn_kernel(const __bf16* __restrict__ Ubf, const __bf16* __restrict__ encbf,
                            const float* __restrict__ proj,
                            const float* __restrict__ v_w, const float* __restrict__ v_b,
                            const float* __restrict__ y_hist, int t,
                            const float* __restrict__ w_w, const float* __restrict__ w_b,
                            float* __restrict__ ctx, float* __restrict__ y_tilde)
{
    __shared__ float sc[128];
    __shared__ float part[4][128];
    float* red = &part[0][0];

    int b    = blockIdx.x;
    int tid  = threadIdx.x;            // 0..127
    int wave = tid >> 5, lane = tid & 31;

    // register-resident slices: e = lane*4 .. lane*4+3
    float4 pr4 = *(const float4*)(proj + (size_t)b * E_ + lane * 4);
    float4 vv4 = *(const float4*)(v_w + lane * 4);

    // ---- pass 1: scores (streams U, bf16, b64 loads) ----
    const __bf16* Ub = Ubf + (size_t)b * T1_ * E_;
    for (int t1 = wave; t1 < T1_; t1 += 4) {
        if (lane < 2)   // prefetch this wave's next row (t1+4), 2 x 128B lines
            __builtin_prefetch((const void*)(Ub + (size_t)(t1 + 4) * E_ + lane * 64), 0, 1);
        bf16x4 u4 = *(const bf16x4*)(Ub + (size_t)t1 * E_ + lane * 4);
        float s = tanhf(pr4.x + (float)u4.x) * vv4.x
                + tanhf(pr4.y + (float)u4.y) * vv4.y
                + tanhf(pr4.z + (float)u4.z) * vv4.z
                + tanhf(pr4.w + (float)u4.w) * vv4.w;
#pragma unroll
        for (int off = 16; off > 0; off >>= 1) s += __shfl_xor(s, off);
        if (lane == 0) sc[t1] = s + v_b[0];
    }
    __syncthreads();

    // ---- softmax over T1 (=127) scores ----
    float x = (tid < T1_) ? sc[tid] : -3.0e38f;
    red[tid] = x;
    __syncthreads();
    for (int s2 = 64; s2 > 0; s2 >>= 1) {
        if (tid < s2) red[tid] = fmaxf(red[tid], red[tid + s2]);
        __syncthreads();
    }
    float mx = red[0];
    __syncthreads();
    float ex = (tid < T1_) ? __expf(x - mx) : 0.f;
    red[tid] = ex;
    __syncthreads();
    for (int s2 = 64; s2 > 0; s2 >>= 1) {
        if (tid < s2) red[tid] += red[tid + s2];
        __syncthreads();
    }
    float inv = 1.f / red[0];
    __syncthreads();
    sc[tid] = ex * inv;                 // beta
    __syncthreads();

    // ---- pass 2: ctx (streams enc, per-wave float4 partials) ----
    const __bf16* eb = encbf + (size_t)b * T1_ * E_;
    float4 acc = {0.f, 0.f, 0.f, 0.f};
    for (int t1 = wave; t1 < T1_; t1 += 4) {
        if (lane < 2)
            __builtin_prefetch((const void*)(eb + (size_t)(t1 + 4) * E_ + lane * 64), 0, 1);
        float beta = sc[t1];
        bf16x4 e4 = *(const bf16x4*)(eb + (size_t)t1 * E_ + lane * 4);
        acc.x += beta * (float)e4.x;
        acc.y += beta * (float)e4.y;
        acc.z += beta * (float)e4.z;
        acc.w += beta * (float)e4.w;
    }
    *(float4*)&part[wave][lane * 4] = acc;
    __syncthreads();

    float cv = part[0][tid] + part[1][tid] + part[2][tid] + part[3][tid];
    ctx[(size_t)b * E_ + tid] = cv;
    __syncthreads();                    // all reads of part done before reuse

    // ---- y_tilde = ctx . w[0:128] + y_t * w[128] + b ----
    red[tid] = cv * w_w[tid];
    __syncthreads();
    for (int s2 = 64; s2 > 0; s2 >>= 1) {
        if (tid < s2) red[tid] += red[tid + s2];
        __syncthreads();
    }
    if (tid == 0)
        y_tilde[b] = red[0] + y_hist[(size_t)b * T1_ + t] * w_w[E_] + w_b[0];
}

__global__ void lstm_update(const float* __restrict__ gates,
                            float* __restrict__ h, float* __restrict__ c,
                            __bf16* __restrict__ h_bf, __bf16* __restrict__ hc_bf)
{
    int idx = blockIdx.x * blockDim.x + threadIdx.x;
    if (idx >= B_ * D_) return;
    int b = idx >> 7, d = idx & 127;
    const float* g = gates + (size_t)b * 4 * D_;
    float ig = g[d], fg = g[D_ + d], gg = g[2 * D_ + d], og = g[3 * D_ + d];
    float cv = c[idx];
    float cn = (1.f / (1.f + __expf(-fg))) * cv
             + (1.f / (1.f + __expf(-ig))) * tanhf(gg);
    float hn = (1.f / (1.f + __expf(-og))) * tanhf(cn);
    h[idx] = hn;  c[idx] = cn;
    h_bf[idx] = (__bf16)hn;
    hc_bf[(size_t)b * 256 + d]       = (__bf16)hn;
    hc_bf[(size_t)b * 256 + 128 + d] = (__bf16)cn;
}

__global__ void prep_kernel(const float* __restrict__ Uw, const float* __restrict__ Wdhs,
                            const float* __restrict__ Whh,
                            const float* __restrict__ bih, const float* __restrict__ bhh,
                            __bf16* __restrict__ Uw_bf, __bf16* __restrict__ Wdhs_bf,
                            __bf16* __restrict__ Whh_bf, float* __restrict__ gbias)
{
    int idx = blockIdx.x * blockDim.x + threadIdx.x;
    if (idx < 128 * 128) Uw_bf[idx]   = (__bf16)Uw[idx];
    if (idx < 128 * 256) Wdhs_bf[idx] = (__bf16)Wdhs[idx];
    if (idx < 512 * 128) Whh_bf[idx]  = (__bf16)Whh[idx];
    if (idx < 512)       gbias[idx]   = bih[idx] + bhh[idx];
}

__global__ void f32_to_bf16(const float* __restrict__ src, __bf16* __restrict__ dst, size_t n) {
    size_t i = (size_t)blockIdx.x * blockDim.x + threadIdx.x;
    if (i < n) dst[i] = (__bf16)src[i];
}

__global__ void init_state(float* __restrict__ h, float* __restrict__ c,
                           __bf16* __restrict__ h_bf, __bf16* __restrict__ hc_bf)
{
    int idx = blockIdx.x * blockDim.x + threadIdx.x;
    if (idx >= B_ * D_) return;
    int b = idx >> 7, d = idx & 127;
    h[idx] = 0.f;  c[idx] = 0.f;
    h_bf[idx] = (__bf16)0.f;
    hc_bf[(size_t)b * 256 + d]       = (__bf16)0.f;
    hc_bf[(size_t)b * 256 + 128 + d] = (__bf16)0.f;
}

__global__ void final_out(const float* __restrict__ h, const float* __restrict__ ctx,
                          const float* __restrict__ fc_w, const float* __restrict__ fc_b,
                          float* __restrict__ out)
{
    __shared__ float red[128];
    int b = blockIdx.x, tid = threadIdx.x;
    red[tid] = h[(size_t)b * 128 + tid] * fc_w[tid]
             + ctx[(size_t)b * 128 + tid] * fc_w[128 + tid];
    __syncthreads();
    for (int s = 64; s > 0; s >>= 1) {
        if (tid < s) red[tid] += red[tid + s];
        __syncthreads();
    }
    if (tid == 0) out[b] = red[0] + fc_b[0];
}

extern "C" void kernel_launch(void* const* d_in, const int* in_sizes, int n_in,
                              void* d_out, int out_size, void* d_ws, size_t ws_size,
                              hipStream_t stream)
{
    const float* enc   = (const float*)d_in[0];
    const float* yhist = (const float*)d_in[1];
    const float* v_w   = (const float*)d_in[2];
    const float* v_b   = (const float*)d_in[3];
    const float* Wdhs  = (const float*)d_in[4];
    const float* bdhs  = (const float*)d_in[5];
    const float* Uw    = (const float*)d_in[6];
    const float* Ubias = (const float*)d_in[7];
    const float* w_w   = (const float*)d_in[8];
    const float* w_b   = (const float*)d_in[9];
    const float* Wih   = (const float*)d_in[10];
    const float* Whh   = (const float*)d_in[11];
    const float* bih   = (const float*)d_in[12];
    const float* bhh   = (const float*)d_in[13];
    const float* fc_w  = (const float*)d_in[14];
    const float* fc_b  = (const float*)d_in[15];
    float* out = (float*)d_out;

    // ---- workspace bump allocator (256B aligned) ----
    char* p = (char*)d_ws;
    auto alloc = [&](size_t bytes) { char* r = p; p += (bytes + 255) & ~(size_t)255; return r; };
    const size_t NE = (size_t)B_ * T1_ * E_;
    __bf16* enc_bf  = (__bf16*)alloc(NE * 2);             // 133 MB
    __bf16* U_bf    = (__bf16*)alloc(NE * 2);             // 133 MB
    __bf16* hc_bf   = (__bf16*)alloc((size_t)B_ * 256 * 2);
    __bf16* h_bf    = (__bf16*)alloc((size_t)B_ * 128 * 2);
    float*  h       = (float*) alloc((size_t)B_ * 128 * 4);
    float*  c       = (float*) alloc((size_t)B_ * 128 * 4);
    float*  proj    = (float*) alloc((size_t)B_ * 128 * 4);
    float*  ctx     = (float*) alloc((size_t)B_ * 128 * 4);
    float*  gates   = (float*) alloc((size_t)B_ * 512 * 4);
    float*  y_tl    = (float*) alloc((size_t)B_ * 4);
    __bf16* Uw_bf   = (__bf16*)alloc(128 * 128 * 2);
    __bf16* Wdhs_bf = (__bf16*)alloc(128 * 256 * 2);
    __bf16* Whh_bf  = (__bf16*)alloc(512 * 128 * 2);
    float*  gbias   = (float*) alloc(512 * 4);

    // ---- one-time prep ----
    prep_kernel<<<256, 256, 0, stream>>>(Uw, Wdhs, Whh, bih, bhh,
                                         Uw_bf, Wdhs_bf, Whh_bf, gbias);
    f32_to_bf16<<<(unsigned)((NE + 255) / 256), 256, 0, stream>>>(enc, enc_bf, NE);
    init_state<<<(B_ * 128) / 256, 256, 0, stream>>>(h, c, h_bf, hc_bf);

    // U = enc @ U_d^T + b : M=520192, N=128, K=128 -> bf16
    {
        int M = B_ * T1_, N = 128, K = 128;
        int waves = (M / 16) * (N / 64);          // 65024
        wmma_gemm4<<<waves / 8, 256, 0, stream>>>(enc_bf, Uw_bf, Ubias,
                                                  nullptr, nullptr, nullptr, U_bf, M, N, K);
    }

    // ---- sequential scan over T1 steps (stream order = dependency) ----
    for (int t = 0; t < T1_; ++t) {
        { // proj = [h,c] @ W_dhs^T + b : M=4096, N=128, K=256
            int M = B_, N = 128, K = 256;
            int waves = (M / 16) * (N / 64);      // 512
            wmma_gemm4<<<waves / 8, 256, 0, stream>>>(hc_bf, Wdhs_bf, bdhs,
                                                      nullptr, nullptr, proj, nullptr, M, N, K);
        }
        attn_kernel<<<B_, 128, 0, stream>>>(U_bf, enc_bf, proj, v_w, v_b,
                                            yhist, t, w_w, w_b, ctx, y_tl);
        { // gates = h @ W_hh^T + (b_ih+b_hh) + y_tilde*W_ih : M=4096, N=512, K=128
            int M = B_, N = 512, K = 128;
            int waves = (M / 16) * (N / 64);      // 2048
            wmma_gemm4<<<waves / 8, 256, 0, stream>>>(h_bf, Whh_bf, gbias,
                                                      y_tl, Wih, gates, nullptr, M, N, K);
        }
        lstm_update<<<(B_ * 128) / 256, 256, 0, stream>>>(gates, h, c, h_bf, hc_bf);
    }

    final_out<<<B_, 128, 0, stream>>>(h, ctx, fc_w, fc_b, out);

    (void)in_sizes; (void)n_in; (void)out_size; (void)ws_size;
}
